// EnGCN_5385888989321
// MI455X (gfx1250) — matrix operand order repaired and checked by reading, not compiled, not run on hardware
//
#include <hip/hip_runtime.h>
#include <hip/hip_bf16.h>

// ---------------------------------------------------------------------------
// EnGCN forward on MI455X (gfx1250, wave32, WMMA)
//   summed = segment_sum(x[src] -> dst); mean = summed / max(deg,1)
//   x1hop = mean @ W_sage
//   out   = (relu(x @ W1a + b1a) @ W1b + b1b) + (relu(x1hop @ W2a + b2a) @ W2b + b2b)
//
// GEMMs: bf16 A/B, f32 accum via v_wmma_f32_16x16x32_bf16.
// A fragments load directly from row-major A (two b128 chunks per lane).
// B fragments load directly from pre-transposed weights Bt[N,K] (contiguous
// 32B per lane) -- no LDS staging, no per-element ds gathers.
// ---------------------------------------------------------------------------

#define IN_FEATS 128
#define HIDDEN   256
#define OUT_FEATS 64

typedef __bf16 v16bf __attribute__((ext_vector_type(16)));
typedef __bf16 v8bf  __attribute__((ext_vector_type(8)));
typedef float  v8f   __attribute__((ext_vector_type(8)));

__device__ __forceinline__ unsigned short f2bf(float f) {
  unsigned u = __builtin_bit_cast(unsigned, f);
  u += 0x7FFFu + ((u >> 16) & 1u);          // round-to-nearest-even
  return (unsigned short)(u >> 16);
}

// ---------------------------------------------------------------- utilities
__global__ void k_zero_f32(float* __restrict__ p, int n) {
  int i = blockIdx.x * blockDim.x + threadIdx.x;
  if (i < n) p[i] = 0.0f;
}

__global__ void k_f32_to_bf16(const float* __restrict__ in,
                              unsigned short* __restrict__ out, int n) {
  int i = blockIdx.x * blockDim.x + threadIdx.x;
  if (i < n) out[i] = f2bf(in[i]);
}

// weights: f32 [K,N] row-major -> bf16 transposed [N,K]
__global__ void k_f32_to_bf16_T(const float* __restrict__ in,
                                unsigned short* __restrict__ out, int K, int N) {
  int i = blockIdx.x * blockDim.x + threadIdx.x;
  if (i >= K * N) return;
  int k = i / N, n = i % N;
  out[(size_t)n * K + k] = f2bf(in[i]);
}

// ------------------------------------------------------- edge scatter (mean)
// One wave32 per edge: 4 feats per lane, hardware f32 atomics into summed.
__global__ void k_scatter(const float* __restrict__ x,
                          const int* __restrict__ ei,
                          float* __restrict__ summed,
                          float* __restrict__ deg,
                          int nEdges, int nNodes) {
  int wave = (blockIdx.x * blockDim.x + threadIdx.x) >> 5;
  int lane = threadIdx.x & 31;
  if (wave >= nEdges) return;
  int src = ei[wave];
  int dst = ei[nEdges + wave];
  if ((unsigned)src >= (unsigned)nNodes || (unsigned)dst >= (unsigned)nNodes) return;
  const float4 v = *(const float4*)(x + (size_t)src * IN_FEATS + lane * 4);
  float* o = summed + (size_t)dst * IN_FEATS + lane * 4;
  __hip_atomic_fetch_add(o + 0, v.x, __ATOMIC_RELAXED, __HIP_MEMORY_SCOPE_AGENT);
  __hip_atomic_fetch_add(o + 1, v.y, __ATOMIC_RELAXED, __HIP_MEMORY_SCOPE_AGENT);
  __hip_atomic_fetch_add(o + 2, v.z, __ATOMIC_RELAXED, __HIP_MEMORY_SCOPE_AGENT);
  __hip_atomic_fetch_add(o + 3, v.w, __ATOMIC_RELAXED, __HIP_MEMORY_SCOPE_AGENT);
  if (lane == 0)
    __hip_atomic_fetch_add(deg + dst, 1.0f, __ATOMIC_RELAXED, __HIP_MEMORY_SCOPE_AGENT);
}

// mean = summed / max(deg,1) -> bf16
__global__ void k_mean_bf16(const float* __restrict__ summed,
                            const float* __restrict__ deg,
                            unsigned short* __restrict__ meanbf, int nNodes) {
  int i = blockIdx.x * blockDim.x + threadIdx.x;
  int total = nNodes * (IN_FEATS / 4);
  if (i >= total) return;
  int node = i >> 5;                 // IN_FEATS/4 == 32
  int c4 = (i & 31) * 4;
  float inv = 1.0f / fmaxf(deg[node], 1.0f);
  const float4 s = *(const float4*)(summed + (size_t)node * IN_FEATS + c4);
  unsigned short* o = meanbf + (size_t)node * IN_FEATS + c4;
  o[0] = f2bf(s.x * inv);
  o[1] = f2bf(s.y * inv);
  o[2] = f2bf(s.z * inv);
  o[3] = f2bf(s.w * inv);
}

// -------------------------------------------------------------- WMMA GEMM
// C[M,N] = act(A[M,K] @ Bt[N,K]^T + bias).  A row-major bf16, Bt = B
// transposed (N x K) bf16.  Block: 256 thr = 8 waves; tile 128(M) x 64(N);
// wave w owns rows w*16..w*16+15 and all 64 columns (4 accumulators).
//
// Fragment layouts as loaded here (contiguous k, wave32, 16-bit):
//   A 16x32: lane l -> row l%16; elems = K[0..15] (+16 upper half)
//            => one contiguous 32B run of the A row.
//   B 32x16: lane l -> col l%16; K sequential (+16 upper half)
//            => one contiguous 32B run of the Bt row.
__global__ void k_wmma_gemm(const unsigned short* __restrict__ A,
                            const unsigned short* __restrict__ Bt,
                            const float* __restrict__ bias,
                            float* __restrict__ Cf,          // f32 out (or null)
                            unsigned short* __restrict__ Cb, // bf16 out (or null)
                            int M, int N, int K,
                            int relu, int accum) {
  const int wave = threadIdx.x >> 5;
  const int lane = threadIdx.x & 31;
  const int hi   = lane >> 4;
  const int l16  = lane & 15;
  const int m0   = blockIdx.x * 128;
  const int n0   = blockIdx.y * 64;

  // Clamp tail rows so every lane issues valid loads and EXEC stays all-ones
  // through the WMMAs; stores are guarded instead.
  int frow = m0 + wave * 16 + l16;
  if (frow >= M) frow = M - 1;
  const __bf16* __restrict__ Ap = (const __bf16*)A + (size_t)frow * K;

  v8f acc0 = {}, acc1 = {}, acc2 = {}, acc3 = {};

  for (int k0 = 0; k0 < K; k0 += 32) {
    // A fragment: contiguous 32B (K = k0+16*hi .. +16)
    v16bf afrag = *(const v16bf*)(Ap + k0 + 16 * hi);

    if (k0 + 32 < K)
      __builtin_prefetch(Ap + k0 + 32, 0, 1);   // global_prefetch_b8

    // B fragments: contiguous 32B per lane from Bt[n, k0+16*hi ..]
    const __bf16* __restrict__ Bp =
        (const __bf16*)Bt + (size_t)(n0 + l16) * K + k0 + 16 * hi;
    v16bf b0 = *(const v16bf*)(Bp);
    v16bf b1 = *(const v16bf*)(Bp + 16 * (size_t)K);
    v16bf b2 = *(const v16bf*)(Bp + 32 * (size_t)K);
    v16bf b3 = *(const v16bf*)(Bp + 48 * (size_t)K);

    acc0 = __builtin_amdgcn_wmma_f32_16x16x32_bf16(false, afrag, false, b0,
                                                   (short)0, acc0, false, false);
    acc1 = __builtin_amdgcn_wmma_f32_16x16x32_bf16(false, afrag, false, b1,
                                                   (short)0, acc1, false, false);
    acc2 = __builtin_amdgcn_wmma_f32_16x16x32_bf16(false, afrag, false, b2,
                                                   (short)0, acc2, false, false);
    acc3 = __builtin_amdgcn_wmma_f32_16x16x32_bf16(false, afrag, false, b3,
                                                   (short)0, acc3, false, false);
  }

  // ---- epilogue: C/D layout = lane holds col n=l16; VGPR v -> row v + 8*hi
  v8f accs[4] = {acc0, acc1, acc2, acc3};
#pragma unroll
  for (int nn = 0; nn < 4; ++nn) {
    const int col = n0 + nn * 16 + l16;
    const float bi = bias ? bias[col] : 0.0f;
#pragma unroll
    for (int v = 0; v < 8; ++v) {
      int row = m0 + wave * 16 + v + 8 * hi;
      if (row >= M) continue;
      float r = accs[nn][v] + bi;
      if (relu) r = fmaxf(r, 0.0f);
      size_t o = (size_t)row * N + col;
      if (Cf) {
        if (accum) r += Cf[o];
        Cf[o] = r;
      } else {
        Cb[o] = f2bf(r);
      }
    }
  }
}

// --------------------------------------------------------------------------
extern "C" void kernel_launch(void* const* d_in, const int* in_sizes, int n_in,
                              void* d_out, int out_size, void* d_ws, size_t ws_size,
                              hipStream_t stream) {
  (void)n_in; (void)out_size; (void)ws_size;

  const float* x      = (const float*)d_in[0];
  const int*   ei     = (const int*)d_in[1];
  const float* W_sage = (const float*)d_in[2];
  const float* W1a    = (const float*)d_in[3];
  const float* b1a    = (const float*)d_in[4];
  const float* W1b    = (const float*)d_in[5];
  const float* b1b    = (const float*)d_in[6];
  const float* W2a    = (const float*)d_in[7];
  const float* b2a    = (const float*)d_in[8];
  const float* W2b    = (const float*)d_in[9];
  const float* b2b    = (const float*)d_in[10];
  float* out = (float*)d_out;

  const int nNodes = in_sizes[0] / IN_FEATS;
  const int nEdges = in_sizes[1] / 2;

  // ---- workspace carve-up (stream-serialized; summed reused as h1_bf16)
  char* ws = (char*)d_ws;
  size_t off = 0;
  auto carve = [&](size_t bytes) {
    char* p = ws + off;
    off = (off + bytes + 255) & ~(size_t)255;
    return p;
  };
  float*          summed  = (float*)carve((size_t)nNodes * IN_FEATS * 4); // also h1 bf16
  float*          deg     = (float*)carve((size_t)nNodes * 4);
  unsigned short* xbf     = (unsigned short*)carve((size_t)nNodes * IN_FEATS * 2);
  unsigned short* meanbf  = (unsigned short*)carve((size_t)nNodes * IN_FEATS * 2);
  unsigned short* x1bf    = (unsigned short*)carve((size_t)nNodes * IN_FEATS * 2);
  unsigned short* h2bf    = (unsigned short*)carve((size_t)nNodes * HIDDEN * 2);
  unsigned short* wsageT  = (unsigned short*)carve((size_t)IN_FEATS * IN_FEATS * 2);
  unsigned short* w1aT    = (unsigned short*)carve((size_t)IN_FEATS * HIDDEN * 2);
  unsigned short* w1bT    = (unsigned short*)carve((size_t)HIDDEN * OUT_FEATS * 2);
  unsigned short* w2aT    = (unsigned short*)carve((size_t)IN_FEATS * HIDDEN * 2);
  unsigned short* w2bT    = (unsigned short*)carve((size_t)HIDDEN * OUT_FEATS * 2);
  unsigned short* h1bf    = (unsigned short*)summed;   // reuse after mean pass

  const int TPB = 256;
  auto blocks = [](long long n, int t) { return (unsigned)((n + t - 1) / t); };

  // 1. zero accumulators
  k_zero_f32<<<blocks((long long)nNodes * IN_FEATS, TPB), TPB, 0, stream>>>(summed, nNodes * IN_FEATS);
  k_zero_f32<<<blocks(nNodes, TPB), TPB, 0, stream>>>(deg, nNodes);

  // 2. edge scatter (one wave32 per edge)
  k_scatter<<<blocks((long long)nEdges * 32, TPB), TPB, 0, stream>>>(x, ei, summed, deg, nEdges, nNodes);

  // 3. mean -> bf16; x -> bf16; weights -> bf16 transposed [N,K]
  k_mean_bf16<<<blocks((long long)nNodes * (IN_FEATS / 4), TPB), TPB, 0, stream>>>(summed, deg, meanbf, nNodes);
  k_f32_to_bf16<<<blocks((long long)nNodes * IN_FEATS, TPB), TPB, 0, stream>>>(x, xbf, nNodes * IN_FEATS);
  k_f32_to_bf16_T<<<blocks(IN_FEATS * IN_FEATS, TPB), TPB, 0, stream>>>(W_sage, wsageT, IN_FEATS, IN_FEATS);
  k_f32_to_bf16_T<<<blocks(IN_FEATS * HIDDEN, TPB), TPB, 0, stream>>>(W1a, w1aT, IN_FEATS, HIDDEN);
  k_f32_to_bf16_T<<<blocks(HIDDEN * OUT_FEATS, TPB), TPB, 0, stream>>>(W1b, w1bT, HIDDEN, OUT_FEATS);
  k_f32_to_bf16_T<<<blocks(IN_FEATS * HIDDEN, TPB), TPB, 0, stream>>>(W2a, w2aT, IN_FEATS, HIDDEN);
  k_f32_to_bf16_T<<<blocks(HIDDEN * OUT_FEATS, TPB), TPB, 0, stream>>>(W2b, w2bT, HIDDEN, OUT_FEATS);

  // 4. WMMA GEMM chain (block tile 128x64)
  const unsigned mB = blocks(nNodes, 128);
  // h1 = relu(x @ W1a + b1a)               [M x 256] bf16   (into summed region)
  k_wmma_gemm<<<dim3(mB, HIDDEN / 64), TPB, 0, stream>>>(xbf, w1aT, b1a, nullptr, h1bf,
                                                         nNodes, HIDDEN, IN_FEATS, 1, 0);
  // x1hop = mean @ W_sage                  [M x 128] bf16
  k_wmma_gemm<<<dim3(mB, IN_FEATS / 64), TPB, 0, stream>>>(meanbf, wsageT, nullptr, nullptr, x1bf,
                                                           nNodes, IN_FEATS, IN_FEATS, 0, 0);
  // h2 = relu(x1hop @ W2a + b2a)           [M x 256] bf16
  k_wmma_gemm<<<dim3(mB, HIDDEN / 64), TPB, 0, stream>>>(x1bf, w2aT, b2a, nullptr, h2bf,
                                                         nNodes, HIDDEN, IN_FEATS, 1, 0);
  // out  = h1 @ W1b + b1b                  [M x 64] f32
  k_wmma_gemm<<<dim3(mB, OUT_FEATS / 64), TPB, 0, stream>>>(h1bf, w1bT, b1b, out, nullptr,
                                                            nNodes, OUT_FEATS, HIDDEN, 0, 0);
  // out += h2 @ W2b + b2b
  k_wmma_gemm<<<dim3(mB, OUT_FEATS / 64), TPB, 0, stream>>>(h2bf, w2bT, b2b, out, nullptr,
                                                            nNodes, OUT_FEATS, HIDDEN, 0, 1);
}
